// MultiheadAttentionZ_83777632076075
// MI455X (gfx1250) — matrix operand order repaired and checked by reading, not compiled, not run on hardware
//
#include <hip/hip_runtime.h>
#include <hip/hip_bf16.h>

// ---------------- problem constants ----------------
#define T_SEQ   2048
#define B_BATCH 2
#define E_DIM   1024
#define H_HEADS 16
#define D_HEAD  64
#define BH_     (B_BATCH * H_HEADS)     // 32
#define M_ROWS  (T_SEQ * B_BATCH)       // 4096

typedef __attribute__((ext_vector_type(16))) _Float16 v16h;
typedef __attribute__((ext_vector_type(8)))  _Float16 v8h;
typedef __attribute__((ext_vector_type(4)))  _Float16 v4h;
typedef __attribute__((ext_vector_type(8)))  float    v8f;
typedef __attribute__((ext_vector_type(4)))  float    v4f;
typedef __attribute__((ext_vector_type(4)))  int      v4i;

// -------- CDNA5 async global->LDS path (guarded; sync staging fallback) -----
#if __has_builtin(__builtin_amdgcn_global_load_async_to_lds_b128) && \
    __has_builtin(__builtin_amdgcn_s_wait_asynccnt)
#define USE_ASYNC_LDS 1
#else
#define USE_ASYNC_LDS 0
#endif

#define AS1 __attribute__((address_space(1)))
#define AS3 __attribute__((address_space(3)))

// copy 16 bytes (8 halves) global -> LDS
__device__ __forceinline__ void copy16B(_Float16* l, const _Float16* g) {
#if USE_ASYNC_LDS
    __builtin_amdgcn_global_load_async_to_lds_b128((AS1 v4i*)g, (AS3 v4i*)l, 0, 0);
#else
    *(v8h*)l = *(const v8h*)g;
#endif
}

__device__ __forceinline__ void async_wait_keep_newest() {
#if USE_ASYNC_LDS
    __builtin_amdgcn_s_wait_asynccnt(4);   // allow only the just-issued stage
#endif
}
__device__ __forceinline__ void async_wait_keep_newest3() {
#if USE_ASYNC_LDS
    __builtin_amdgcn_s_wait_asynccnt(3);
#endif
}
__device__ __forceinline__ void async_wait_all() {
#if USE_ASYNC_LDS
    __builtin_amdgcn_s_wait_asynccnt(0);
#endif
}

__device__ __forceinline__ v8f v8f_zero() {
    v8f z = {0.f, 0.f, 0.f, 0.f, 0.f, 0.f, 0.f, 0.f};
    return z;
}

// A-fragment (16x32 f16, MxK): lane L holds row M=L%16;
// lanes 0-15: K = {0..7, 16..23}, lanes 16-31: K = {8..15, 24..31}
__device__ __forceinline__ v16h load_frag_a(const _Float16* __restrict__ base, int stride) {
    const int lane = threadIdx.x & 31;
    const int r  = lane & 15;
    const int ko = (lane >> 4) << 3;           // 0 or 8
    const v8h c0 = *(const v8h*)(base + r * stride + ko);
    const v8h c1 = *(const v8h*)(base + r * stride + ko + 16);
    v16h f;
#pragma unroll
    for (int i = 0; i < 8; ++i) { f[i] = c0[i]; f[i + 8] = c1[i]; }
    return f;
}

// B-fragment (32x16 f16, KxN), stored here as B^T rows (N rows, K contiguous):
// lane L holds column N=L%16; lanes 0-15: K = {0..7, 16..23}, lanes 16-31: K = {8..15, 24..31}
__device__ __forceinline__ v16h load_frag_b(const _Float16* __restrict__ base, int stride) {
    const int lane = threadIdx.x & 31;
    const int n  = lane & 15;
    const int kb = (lane >> 4) << 3;           // 0 or 8
    const v8h c0 = *(const v8h*)(base + n * stride + kb);
    const v8h c1 = *(const v8h*)(base + n * stride + kb + 16);
    v16h f;
#pragma unroll
    for (int i = 0; i < 8; ++i) { f[i] = c0[i]; f[i + 8] = c1[i]; }
    return f;
}

// ---------------- fp32 -> f16 convert ----------------
__global__ __launch_bounds__(256)
void cvt_f16_kernel(const float* __restrict__ src, _Float16* __restrict__ dst, int n4) {
    int i = blockIdx.x * 256 + threadIdx.x;
    if (i < n4) {
        v4f v = *(const v4f*)(src + (size_t)i * 4);
        v4h o;
#pragma unroll
        for (int e = 0; e < 4; ++e) o[e] = (_Float16)v[e];
        *(v4h*)(dst + (size_t)i * 4) = o;
    }
}

// ---------------- generic 128x128 WMMA GEMM (double-buffered LDS) -----------
// C = A(MxK, row-major, lda) * B^T where Bw is (N x K, row-major, ldb), + bias.
// MODE 0: q-proj  -> f16, head layout (bh,t,d), scaled by 1/8
// MODE 1: k-proj  -> f16, head layout (bh,t,d)
// MODE 2: v-proj  -> f16, transposed head layout (bh,d,t)
// MODE 3: out-proj-> f32, row-major (M x N)
// MODE 4: scores  -> f16, row-major (M x N), batched via blockIdx.z
template <int MODE>
__global__ __launch_bounds__(256)
void gemm128_kernel(const _Float16* __restrict__ A, const _Float16* __restrict__ Bw,
                    const float* __restrict__ bias, void* __restrict__ dstv,
                    int M, int N, int K, int lda, int ldb,
                    size_t bsA, size_t bsB, size_t bsD) {
    __shared__ __align__(16) _Float16 As[2][128][40];
    __shared__ __align__(16) _Float16 Bs[2][128][40];

    const int tid  = threadIdx.x;
    const int wave = tid >> 5;
    const int z    = blockIdx.z;
    const int bm   = blockIdx.y * 128;
    const int bn   = blockIdx.x * 128;
    // 8 waves as 4x2; each wave computes 32 rows x 64 cols (2x4 WMMA frags)
    const int wm = (wave >> 1) * 32;
    const int wn = (wave & 1) * 64;

    A  += z * bsA;
    Bw += z * bsB;

    // per-thread staging coordinates: 512 16B chunks per 128x32 tile, 2/thread
    const int r0  = tid >> 2,          ko0 = (tid & 3) << 3;
    const int r1  = (tid + 256) >> 2,  ko1 = ((tid + 256) & 3) << 3;

    v8f acc[2][4];
#pragma unroll
    for (int i = 0; i < 2; ++i)
#pragma unroll
        for (int j = 0; j < 4; ++j) acc[i][j] = v8f_zero();

    // prologue: stage K-tile 0 into buffer 0
    copy16B(&As[0][r0][ko0], A  + (size_t)(bm + r0) * lda + ko0);
    copy16B(&Bs[0][r0][ko0], Bw + (size_t)(bn + r0) * ldb + ko0);
    copy16B(&As[0][r1][ko1], A  + (size_t)(bm + r1) * lda + ko1);
    copy16B(&Bs[0][r1][ko1], Bw + (size_t)(bn + r1) * ldb + ko1);

    for (int k0 = 0; k0 < K; k0 += 32) {
        const int cur = (k0 >> 5) & 1;
        const int nxt = cur ^ 1;
        if (k0 + 32 < K) {   // stage next tile while computing this one
            const int kn = k0 + 32;
            copy16B(&As[nxt][r0][ko0], A  + (size_t)(bm + r0) * lda + kn + ko0);
            copy16B(&Bs[nxt][r0][ko0], Bw + (size_t)(bn + r0) * ldb + kn + ko0);
            copy16B(&As[nxt][r1][ko1], A  + (size_t)(bm + r1) * lda + kn + ko1);
            copy16B(&Bs[nxt][r1][ko1], Bw + (size_t)(bn + r1) * ldb + kn + ko1);
            if (k0 + 64 < K)  // long-range prefetch into L2
                __builtin_prefetch(A + (size_t)(bm + (tid >> 1)) * lda + k0 + 64 +
                                       ((tid & 1) << 4), 0, 0);
            async_wait_keep_newest();
        } else {
            async_wait_all();
        }
        __syncthreads();   // current buffer visible to all waves

        v16h af[2], bf[4];
#pragma unroll
        for (int i = 0; i < 2; ++i) af[i] = load_frag_a(&As[cur][wm + i * 16][0], 40);
#pragma unroll
        for (int j = 0; j < 4; ++j) bf[j] = load_frag_b(&Bs[cur][wn + j * 16][0], 40);
#pragma unroll
        for (int i = 0; i < 2; ++i)
#pragma unroll
            for (int j = 0; j < 4; ++j)
                acc[i][j] = __builtin_amdgcn_wmma_f32_16x16x32_f16(
                    false, af[i], false, bf[j], (short)0, acc[i][j], false, false);
        __syncthreads();   // all waves done with 'cur' before it is restaged
    }

    // epilogue: C/D layout -> m = (lane>>4)*8 + vgpr, n = lane&15
    const int lane = tid & 31;
    const int nl = lane & 15;
    const int m0 = (lane >> 4) << 3;
#pragma unroll
    for (int i = 0; i < 2; ++i) {
#pragma unroll
        for (int j = 0; j < 4; ++j) {
            const int gm_base = bm + wm + i * 16 + m0;
            const int gn      = bn + wn + j * 16 + nl;
            const float bv = (MODE == 4) ? 0.0f : bias[gn];
#pragma unroll
            for (int e = 0; e < 8; ++e) {
                const int gm = gm_base + e;
                float val = acc[i][j][e] + bv;
                if (MODE == 3) {
                    ((float*)dstv)[(size_t)gm * N + gn] = val;
                } else if (MODE == 4) {
                    ((_Float16*)dstv + z * bsD)[(size_t)gm * N + gn] = (_Float16)val;
                } else {
                    const int t = gm / B_BATCH, b = gm % B_BATCH;
                    const int h = gn >> 6, dd = gn & 63;
                    const int bh = b * H_HEADS + h;
                    if (MODE == 0) val *= 0.125f;  // 1/sqrt(64)
                    const _Float16 hv = (_Float16)val;
                    if (MODE == 2)
                        ((_Float16*)dstv)[((size_t)bh * D_HEAD + dd) * T_SEQ + t] = hv;
                    else
                        ((_Float16*)dstv)[((size_t)bh * T_SEQ + t) * D_HEAD + dd] = hv;
                }
            }
        }
    }
}

// ---------------- softmax over rows of (BH, T, S), f16 in-place, fp32 math ----
__global__ __launch_bounds__(256)
void softmax_kernel(_Float16* __restrict__ sc) {
    const int t = blockIdx.x;
    const int z = blockIdx.y;
    _Float16* row = sc + ((size_t)z * T_SEQ + t) * T_SEQ;
    __shared__ float red[256];
    const int tid = threadIdx.x;

    float vals[8];
    float m = -1e30f;
#pragma unroll
    for (int i = 0; i < 8; ++i) {
        vals[i] = (float)row[tid + i * 256];
        m = fmaxf(m, vals[i]);
    }
    red[tid] = m;
    __syncthreads();
    for (int s = 128; s > 0; s >>= 1) {
        if (tid < s) red[tid] = fmaxf(red[tid], red[tid + s]);
        __syncthreads();
    }
    m = red[0];
    __syncthreads();

    float sum = 0.f;
#pragma unroll
    for (int i = 0; i < 8; ++i) { vals[i] = __expf(vals[i] - m); sum += vals[i]; }
    red[tid] = sum;
    __syncthreads();
    for (int s = 128; s > 0; s >>= 1) {
        if (tid < s) red[tid] += red[tid + s];
        __syncthreads();
    }
    const float inv = 1.0f / red[0];
#pragma unroll
    for (int i = 0; i < 8; ++i) row[tid + i * 256] = (_Float16)(vals[i] * inv);
}

// ---------------- attn_avg[b,t,s] = (1/H) sum_h attn[(b*H+h),t,s] ------------
__global__ __launch_bounds__(256)
void attn_avg_kernel(const _Float16* __restrict__ attn, float* __restrict__ out2) {
    const int t = blockIdx.x;
    const int b = blockIdx.y;
    const size_t base = (size_t)b * H_HEADS * T_SEQ * T_SEQ + (size_t)t * T_SEQ;
    float* dstrow = out2 + ((size_t)b * T_SEQ + t) * T_SEQ;
    for (int s = threadIdx.x; s < T_SEQ; s += 256) {
        float acc = 0.f;
#pragma unroll
        for (int h = 0; h < H_HEADS; ++h)
            acc += (float)attn[base + (size_t)h * T_SEQ * T_SEQ + s];
        dstrow[s] = acc * (1.0f / H_HEADS);
    }
}

// ---------------- AV GEMM: per head, out(T x 64) = attn(T x S) * v(S x 64) ---
// B operand supplied as v^T (64 rows x T, K-contiguous). Double-buffered LDS.
__global__ __launch_bounds__(256)
void gemm_av_kernel(const _Float16* __restrict__ attn, const _Float16* __restrict__ vT,
                    _Float16* __restrict__ aout) {
    __shared__ __align__(16) _Float16 As[2][128][40];
    __shared__ __align__(16) _Float16 Bs[2][64][40];

    const int tid  = threadIdx.x;
    const int wave = tid >> 5;
    const int z    = blockIdx.z;               // bh
    const int bm   = blockIdx.y * 128;         // t tile
    const int wm = (wave >> 1) * 32;
    const int wn = (wave & 1) * 32;            // 8 waves as 4x2 over 128x64

    const _Float16* Ab = attn + (size_t)z * T_SEQ * T_SEQ;
    const _Float16* Bb = vT   + (size_t)z * D_HEAD * T_SEQ;

    const int r0  = tid >> 2,          ko0 = (tid & 3) << 3;
    const int r1  = (tid + 256) >> 2,  ko1 = ((tid + 256) & 3) << 3;
    const int rb  = tid >> 2;          // 64 rows x 4 chunks = 256 per B tile

    v8f acc[2][2];
#pragma unroll
    for (int i = 0; i < 2; ++i)
#pragma unroll
        for (int j = 0; j < 2; ++j) acc[i][j] = v8f_zero();

    copy16B(&As[0][r0][ko0], Ab + (size_t)(bm + r0) * T_SEQ + ko0);
    copy16B(&As[0][r1][ko1], Ab + (size_t)(bm + r1) * T_SEQ + ko1);
    copy16B(&Bs[0][rb][ko0], Bb + (size_t)rb * T_SEQ + ko0);

    for (int k0 = 0; k0 < T_SEQ; k0 += 32) {
        const int cur = (k0 >> 5) & 1;
        const int nxt = cur ^ 1;
        if (k0 + 32 < T_SEQ) {
            const int kn = k0 + 32;
            copy16B(&As[nxt][r0][ko0], Ab + (size_t)(bm + r0) * T_SEQ + kn + ko0);
            copy16B(&As[nxt][r1][ko1], Ab + (size_t)(bm + r1) * T_SEQ + kn + ko1);
            copy16B(&Bs[nxt][rb][ko0], Bb + (size_t)rb * T_SEQ + kn + ko0);
            if (k0 + 64 < T_SEQ)
                __builtin_prefetch(Ab + (size_t)(bm + (tid >> 1)) * T_SEQ + k0 + 64 +
                                       ((tid & 1) << 4), 0, 0);
            async_wait_keep_newest3();
        } else {
            async_wait_all();
        }
        __syncthreads();

        v16h af[2], bf[2];
#pragma unroll
        for (int i = 0; i < 2; ++i) af[i] = load_frag_a(&As[cur][wm + i * 16][0], 40);
#pragma unroll
        for (int j = 0; j < 2; ++j) bf[j] = load_frag_b(&Bs[cur][wn + j * 16][0], 40);
#pragma unroll
        for (int i = 0; i < 2; ++i)
#pragma unroll
            for (int j = 0; j < 2; ++j)
                acc[i][j] = __builtin_amdgcn_wmma_f32_16x16x32_f16(
                    false, af[i], false, bf[j], (short)0, acc[i][j], false, false);
        __syncthreads();
    }

    const int lane = tid & 31;
    const int nl = lane & 15;
    const int m0 = (lane >> 4) << 3;
    const int b = z >> 4, h = z & 15;
#pragma unroll
    for (int i = 0; i < 2; ++i) {
#pragma unroll
        for (int j = 0; j < 2; ++j) {
#pragma unroll
            for (int e = 0; e < 8; ++e) {
                const int t  = bm + wm + i * 16 + m0 + e;
                const int dd = wn + j * 16 + nl;
                aout[((size_t)t * B_BATCH + b) * E_DIM + h * D_HEAD + dd] =
                    (_Float16)acc[i][j][e];
            }
        }
    }
}

// ---------------- host launcher ----------------
extern "C" void kernel_launch(void* const* d_in, const int* in_sizes, int n_in,
                              void* d_out, int out_size, void* d_ws, size_t ws_size,
                              hipStream_t stream) {
    (void)in_sizes; (void)n_in; (void)out_size; (void)ws_size;
    const float* query = (const float*)d_in[0];
    const float* key_  = (const float*)d_in[1];
    const float* value = (const float*)d_in[2];
    const float* w_in  = (const float*)d_in[3];
    const float* b_in  = (const float*)d_in[4];
    const float* w_out = (const float*)d_in[5];
    const float* b_out = (const float*)d_in[6];

    float* out      = (float*)d_out;                      // (T,B,E) = 4096x1024
    float* attn_avg = out + (size_t)M_ROWS * E_DIM;       // (B,T,S)

    // workspace layout (bytes)
    char* ws = (char*)d_ws;
    _Float16* xq   = (_Float16*)(ws + (size_t) 0  * (1 << 20));  // 8 MB each
    _Float16* xk   = (_Float16*)(ws + (size_t) 8  * (1 << 20));
    _Float16* xv   = (_Float16*)(ws + (size_t)16  * (1 << 20));
    _Float16* wi   = (_Float16*)(ws + (size_t)24  * (1 << 20));  // 6 MB
    _Float16* wo   = (_Float16*)(ws + (size_t)30  * (1 << 20));  // 2 MB
    _Float16* qh   = (_Float16*)(ws + (size_t)32  * (1 << 20));  // 8 MB
    _Float16* kh   = (_Float16*)(ws + (size_t)40  * (1 << 20));  // 8 MB
    _Float16* vT   = (_Float16*)(ws + (size_t)48  * (1 << 20));  // 8 MB
    _Float16* aout = (_Float16*)(ws + (size_t)56  * (1 << 20));  // 8 MB
    _Float16* sc   = (_Float16*)(ws + (size_t)64  * (1 << 20));  // 256 MB (scores/attn)

    const int nx = M_ROWS * E_DIM;            // 4,194,304
    const int nwi = 3 * E_DIM * E_DIM;        // 3,145,728
    const int nwo = E_DIM * E_DIM;            // 1,048,576
    cvt_f16_kernel<<<nx / 1024,  256, 0, stream>>>(query, xq, nx / 4);
    cvt_f16_kernel<<<nx / 1024,  256, 0, stream>>>(key_,  xk, nx / 4);
    cvt_f16_kernel<<<nx / 1024,  256, 0, stream>>>(value, xv, nx / 4);
    cvt_f16_kernel<<<nwi / 1024, 256, 0, stream>>>(w_in,  wi, nwi / 4);
    cvt_f16_kernel<<<nwo / 1024, 256, 0, stream>>>(w_out, wo, nwo / 4);

    // in-projections: (4096x1024) = x * Wsec^T + bsec
    dim3 gproj(E_DIM / 128, M_ROWS / 128, 1);  // (8, 32)
    gemm128_kernel<0><<<gproj, 256, 0, stream>>>(xq, wi, b_in, qh,
        M_ROWS, E_DIM, E_DIM, E_DIM, E_DIM, 0, 0, 0);
    gemm128_kernel<1><<<gproj, 256, 0, stream>>>(xk, wi + (size_t)E_DIM * E_DIM, b_in + E_DIM, kh,
        M_ROWS, E_DIM, E_DIM, E_DIM, E_DIM, 0, 0, 0);
    gemm128_kernel<2><<<gproj, 256, 0, stream>>>(xv, wi + (size_t)2 * E_DIM * E_DIM, b_in + 2 * E_DIM, vT,
        M_ROWS, E_DIM, E_DIM, E_DIM, E_DIM, 0, 0, 0);

    // scores: per head (T x S), K = 64
    dim3 gsc(T_SEQ / 128, T_SEQ / 128, BH_);   // (16, 16, 32)
    gemm128_kernel<4><<<gsc, 256, 0, stream>>>(qh, kh, nullptr, sc,
        T_SEQ, T_SEQ, D_HEAD, D_HEAD, D_HEAD,
        (size_t)T_SEQ * D_HEAD, (size_t)T_SEQ * D_HEAD, (size_t)T_SEQ * T_SEQ);

    // softmax in-place, then head-average
    softmax_kernel<<<dim3(T_SEQ, BH_), 256, 0, stream>>>(sc);
    attn_avg_kernel<<<dim3(T_SEQ, B_BATCH), 256, 0, stream>>>(sc, attn_avg);

    // attn @ v -> (T*B, E) f16
    gemm_av_kernel<<<dim3(1, T_SEQ / 128, BH_), 256, 0, stream>>>(sc, vT, aout);

    // out-projection -> fp32 d_out
    gemm128_kernel<3><<<gproj, 256, 0, stream>>>(aout, wo, b_out, out,
        M_ROWS, E_DIM, E_DIM, E_DIM, E_DIM, 0, 0, 0);
}
